// ChildSumTreeLSTMCell_22351009808691
// MI455X (gfx1250) — compile-verified, matrix operand order checked by reading, not run on hardware
//
#include <hip/hip_runtime.h>
#include <stdint.h>

typedef __attribute__((ext_vector_type(16))) __bf16 v16bf;
typedef __attribute__((ext_vector_type(8)))  float  v8f;

#define HS   256
#define XS   300
#define XSP  320          // X padded to multiple of 32 (zeros)
#define KC   4            // children
#define MT   32           // nodes per block
#define NTOT 65536

// ---------- helpers ----------
__device__ __forceinline__ unsigned short f2bf(float f) {
  unsigned u = __builtin_bit_cast(unsigned, f);
  unsigned r = u + 0x7FFFu + ((u >> 16) & 1u);   // round-to-nearest-even
  return (unsigned short)(r >> 16);
}
__device__ __forceinline__ float bf2f(unsigned short b) {
  unsigned u = ((unsigned)b) << 16;
  return __builtin_bit_cast(float, u);
}
__device__ __forceinline__ float sigmoidf_(float x) {
  return 1.0f / (1.0f + __expf(-x));
}

union FragU { struct { uint4 lo, hi; } q; v16bf v; };

// base points at element (row*ld + k0 + kh); chunks at +0 and +16 elements.
__device__ __forceinline__ v16bf load_frag(const unsigned short* base) {
  FragU f;
  f.q.lo = *(const uint4*)(base);
  f.q.hi = *(const uint4*)(base + 16);
  return f.v;
}

__device__ __forceinline__ v8f vzero8() {
  v8f z;
#pragma unroll
  for (int r = 0; r < 8; ++r) z[r] = 0.0f;
  return z;
}

// ---------- prep: fp32 weights -> bf16 packed in workspace ----------
// WxG: [1024][320]  rows 0..255 = W_f, rows 256..1023 = W_iou (cols>=300 zero)
// UfG: [256][256], UiG: [768][256]
__global__ void treelstm_prep(const float* __restrict__ W_iou,
                              const float* __restrict__ U_iou,
                              const float* __restrict__ W_f,
                              const float* __restrict__ U_f,
                              unsigned short* __restrict__ WxG,
                              unsigned short* __restrict__ UfG,
                              unsigned short* __restrict__ UiG) {
  int i = blockIdx.x * blockDim.x + threadIdx.x;
  if (i < 1024 * XSP) {
    int r = i / XSP, c = i % XSP;
    float v = 0.0f;
    if (c < XS) v = (r < 256) ? W_f[r * XS + c] : W_iou[(r - 256) * XS + c];
    WxG[i] = f2bf(v);
  }
  if (i < 256 * HS) UfG[i] = f2bf(U_f[i]);
  if (i < 768 * HS) UiG[i] = f2bf(U_iou[i]);
}

// ---------- fused main kernel: one pass per 32-node tile ----------
// LDS layout (bytes):
//   xs   [32][320] bf16   @      0  (20480)
//   hsm  [128][256] bf16  @  20480  (65536)
//   htl  [32][256] bf16   @  86016  (16384)
//   wxf  [32][256] f32    @ 102400  (32768)
//   iou  [32][768] f32    @ 135168  (98304)
//   ctl  [32][256] f32    @ 233472  (32768)   total 266240
__global__ __launch_bounds__(256)
void treelstm_main(const float* __restrict__ x,
                   const float* __restrict__ h_msgs,
                   const float* __restrict__ c_msgs,
                   const float* __restrict__ b_iou,
                   const float* __restrict__ b_Uiou,
                   const float* __restrict__ b_Wf,
                   const float* __restrict__ b_Uf,
                   const unsigned short* __restrict__ WxG,
                   const unsigned short* __restrict__ UfG,
                   const unsigned short* __restrict__ UiG,
                   float* __restrict__ out) {
  extern __shared__ char smem[];
  unsigned short* xs  = (unsigned short*)(smem);
  unsigned short* hsm = (unsigned short*)(smem + 20480);
  unsigned short* htl = (unsigned short*)(smem + 86016);
  float* wxf = (float*)(smem + 102400);
  float* iou = (float*)(smem + 135168);
  float* ctl = (float*)(smem + 233472);

  const int tid = threadIdx.x;
  const long long n0 = (long long)blockIdx.x * MT;

  // ---- stage x tile (bf16, zero-padded to 320) ----
  {
    int r0 = tid >> 4;       // 0..15
    int c0 = tid & 15;
#pragma unroll
    for (int rr = 0; rr < 2; ++rr) {
      int r = r0 + rr * 16;
      const float* xp = x + (n0 + r) * XS;
      for (int c = c0; c < XS; c += 16) xs[r * XSP + c] = f2bf(xp[c]);
      for (int c = XS + c0; c < XSP; c += 16) xs[r * XSP + c] = 0;
    }
  }
  // ---- stage h_msgs tile: 128 rows (node*4+k) x 256, bf16 ----
  {
    int r  = tid >> 1;       // 0..127
    int c0 = (tid & 1) * 128;
    const float4* hp = (const float4*)(h_msgs + (n0 * KC + r) * HS);
#pragma unroll
    for (int j = 0; j < 32; ++j) {
      float4 v = hp[(c0 >> 2) + j];
      uint2 pk;
      pk.x = (unsigned)f2bf(v.x) | ((unsigned)f2bf(v.y) << 16);
      pk.y = (unsigned)f2bf(v.z) | ((unsigned)f2bf(v.w) << 16);
      *(uint2*)(hsm + r * HS + c0 + j * 4) = pk;
    }
  }
  __syncthreads();

  // ---- h_tild = sum_k h_msgs (bf16 in LDS) ----
  {
    int n  = tid >> 3;            // 0..31
    int c0 = (tid & 7) * 32;
#pragma unroll
    for (int j = 0; j < 32; ++j) {
      int c = c0 + j;
      float s = bf2f(hsm[(n * 4 + 0) * HS + c]) + bf2f(hsm[(n * 4 + 1) * HS + c]) +
                bf2f(hsm[(n * 4 + 2) * HS + c]) + bf2f(hsm[(n * 4 + 3) * HS + c]);
      htl[n * HS + c] = f2bf(s);
    }
  }
  __syncthreads();

  const int wave = tid >> 5;
  const int lane = tid & 31;
  const int m    = lane & 15;
  const int kh   = (lane >> 4) << 3;   // K half select (A/B frag layout)
  const int mhi  = (lane >> 4) << 3;   // C/D layout: M = r + mhi

  // ---- GEMM1: x[32x320] @ WxG.T ; wave owns cols [wave*128, wave*128+128) ----
  v8f acc1[2][8];
#pragma unroll
  for (int rt = 0; rt < 2; ++rt)
#pragma unroll
    for (int t = 0; t < 8; ++t) acc1[rt][t] = vzero8();

  for (int k0 = 0; k0 < XSP; k0 += 32) {
    v16bf a0 = load_frag(xs + (m)      * XSP + k0 + kh);
    v16bf a1 = load_frag(xs + (16 + m) * XSP + k0 + kh);
#pragma unroll
    for (int t = 0; t < 8; ++t) {
      v16bf b = load_frag(WxG + (wave * 128 + t * 16 + m) * XSP + k0 + kh);
      acc1[0][t] = __builtin_amdgcn_wmma_f32_16x16x32_bf16(
          false, a0, false, b, (short)0, acc1[0][t], false, false);
      acc1[1][t] = __builtin_amdgcn_wmma_f32_16x16x32_bf16(
          false, a1, false, b, (short)0, acc1[1][t], false, false);
    }
  }
  if (wave < 2) {  // cols 0..255 = x@W_f.T -> wxf LDS
#pragma unroll
    for (int t = 0; t < 8; ++t) {
      int col = wave * 128 + t * 16 + m;
#pragma unroll
      for (int rt = 0; rt < 2; ++rt)
#pragma unroll
        for (int r = 0; r < 8; ++r)
          wxf[(rt * 16 + r + mhi) * HS + col] = acc1[rt][t][r];
    }
  }

  // ---- GEMM3: iou += h_tild[32x256] @ U_iou.T (merged into acc1, waves 2..7) ----
  if (wave >= 2) {
    int gb = wave * 128 - 256;   // 0..640
    for (int k0 = 0; k0 < HS; k0 += 32) {
      v16bf a0 = load_frag(htl + (m)      * HS + k0 + kh);
      v16bf a1 = load_frag(htl + (16 + m) * HS + k0 + kh);
#pragma unroll
      for (int t = 0; t < 8; ++t) {
        v16bf b = load_frag(UiG + (gb + t * 16 + m) * HS + k0 + kh);
        acc1[0][t] = __builtin_amdgcn_wmma_f32_16x16x32_bf16(
            false, a0, false, b, (short)0, acc1[0][t], false, false);
        acc1[1][t] = __builtin_amdgcn_wmma_f32_16x16x32_bf16(
            false, a1, false, b, (short)0, acc1[1][t], false, false);
      }
    }
#pragma unroll
    for (int t = 0; t < 8; ++t) {
      int col = gb + t * 16 + m;
#pragma unroll
      for (int rt = 0; rt < 2; ++rt)
#pragma unroll
        for (int r = 0; r < 8; ++r)
          iou[(rt * 16 + r + mhi) * 768 + col] = acc1[rt][t][r];
    }
  }
  __syncthreads();   // wxf + iou ready; acc1 dead

  // ---- GEMM2: uh = h_flat[128x256] @ U_f.T, fused f-gate + c_tild epilogue ----
  // wave owns 2 unique col tiles (cols wave*32 .. wave*32+31), all 8 row tiles.
  {
    v8f acc2[8][2];
#pragma unroll
    for (int rt = 0; rt < 8; ++rt)
#pragma unroll
      for (int ct = 0; ct < 2; ++ct) acc2[rt][ct] = vzero8();

    const int cbase = wave * 32;
    for (int k0 = 0; k0 < HS; k0 += 32) {
      v16bf b0 = load_frag(UfG + (cbase + m)      * HS + k0 + kh);
      v16bf b1 = load_frag(UfG + (cbase + 16 + m) * HS + k0 + kh);
#pragma unroll
      for (int rt = 0; rt < 8; ++rt) {
        v16bf a = load_frag(hsm + (rt * 16 + m) * HS + k0 + kh);
        acc2[rt][0] = __builtin_amdgcn_wmma_f32_16x16x32_bf16(
            false, a, false, b0, (short)0, acc2[rt][0], false, false);
        acc2[rt][1] = __builtin_amdgcn_wmma_f32_16x16x32_bf16(
            false, a, false, b1, (short)0, acc2[rt][1], false, false);
      }
    }

    // epilogue: C-tile rows rt*16 + mhi + r  ->  node = rt*4 + (mhi>>2) + (r>=4),
    // k = r&3.  Each lane locally reduces over k and owns unique (node,col).
#pragma unroll
    for (int ct = 0; ct < 2; ++ct) {
      int col = cbase + ct * 16 + m;
      float bw = b_Wf[col] + b_Uf[col];
#pragma unroll
      for (int rt = 0; rt < 8; ++rt) {
        int node0 = rt * 4 + (mhi >> 2);
        float wx0 = wxf[node0 * HS + col] + bw;
        float wx1 = wxf[(node0 + 1) * HS + col] + bw;
        const float* cm0 = c_msgs + (n0 + node0) * KC * HS + col;
        const float* cm1 = c_msgs + (n0 + node0 + 1) * KC * HS + col;
        float s0 = 0.0f, s1 = 0.0f;
#pragma unroll
        for (int r = 0; r < 4; ++r) {
          s0 += sigmoidf_(wx0 + acc2[rt][ct][r])     * cm0[r * HS];
          s1 += sigmoidf_(wx1 + acc2[rt][ct][r + 4]) * cm1[r * HS];
        }
        ctl[node0 * HS + col]       = s0;
        ctl[(node0 + 1) * HS + col] = s1;
      }
    }
  }
  __syncthreads();

  // ---- final gates: thread tid owns column h = tid ----
  {
    int h = tid;
    float bi = b_iou[h]       + b_Uiou[h];
    float bo = b_iou[256 + h] + b_Uiou[256 + h];
    float bu = b_iou[512 + h] + b_Uiou[512 + h];
    for (int n = 0; n < MT; ++n) {
      float iv = sigmoidf_(iou[n * 768 + h] + bi);
      float ov = sigmoidf_(iou[n * 768 + 256 + h] + bo);
      float uv = tanhf(iou[n * 768 + 512 + h] + bu);
      float c  = iv * uv + ctl[n * HS + h];
      float hv = ov * tanhf(c);
      long long row = n0 + n;
      out[row * HS + h] = hv;
      out[(long long)NTOT * HS + row * HS + h] = c;
    }
  }
}

// ---------- launch ----------
extern "C" void kernel_launch(void* const* d_in, const int* in_sizes, int n_in,
                              void* d_out, int out_size, void* d_ws, size_t ws_size,
                              hipStream_t stream) {
  (void)in_sizes; (void)n_in; (void)out_size; (void)ws_size;
  const float* x      = (const float*)d_in[0];
  const float* h_msgs = (const float*)d_in[1];
  const float* c_msgs = (const float*)d_in[2];
  const float* W_iou  = (const float*)d_in[3];
  const float* b_iou  = (const float*)d_in[4];
  const float* U_iou  = (const float*)d_in[5];
  const float* b_Uiou = (const float*)d_in[6];
  const float* W_f    = (const float*)d_in[7];
  const float* b_Wf   = (const float*)d_in[8];
  const float* U_f    = (const float*)d_in[9];
  const float* b_Uf   = (const float*)d_in[10];

  unsigned short* WxG = (unsigned short*)d_ws;     // 1024*320 bf16
  unsigned short* UfG = WxG + 1024 * XSP;          // 256*256 bf16
  unsigned short* UiG = UfG + 256 * HS;            // 768*256 bf16

  treelstm_prep<<<(1024 * XSP) / 256, 256, 0, stream>>>(W_iou, U_iou, W_f, U_f,
                                                        WxG, UfG, UiG);

  const size_t smem_bytes = 266240;  // 260 KB < 320 KB/WGP
  treelstm_main<<<NTOT / MT, 256, smem_bytes, stream>>>(
      x, h_msgs, c_msgs, b_iou, b_Uiou, b_Wf, b_Uf, WxG, UfG, UiG, (float*)d_out);
}
